// TICSContrastiveLoss_38835094290678
// MI455X (gfx1250) — matrix operand order repaired and checked by reading, not compiled, not run on hardware
//
#include <hip/hip_runtime.h>

typedef __attribute__((ext_vector_type(16))) __bf16 v16bf;
typedef __attribute__((ext_vector_type(2)))  __bf16 v2bf;
typedef __attribute__((ext_vector_type(8)))  float  v8f;
typedef __attribute__((ext_vector_type(8)))  unsigned v8u;

#define T_DIM 2048
#define B_DIM 16
#define D_DIM 256
#define ROWS_PER_BLOCK 128
#define NBLK (T_DIM / ROWS_PER_BLOCK)
// Work in base-2: fold log2(e) into the 1/temperature scale so the online
// softmax uses bare v_exp_f32 / v_log_f32 (which are natively base-2).
#define SCALE2 14.4269504088896340736f   // 10.0 * log2(e)
#define LN2    0.69314718055994530942f
#define NEGINF -1e30f

__device__ __forceinline__ float exp2_fast(float x) { return __builtin_amdgcn_exp2f(x); }
__device__ __forceinline__ float log2_fast(float x) { return __builtin_amdgcn_logf(x); }

// fp32 -> packed bf16x2 via compiler fptrunc (backend picks native cvt on gfx1250).
__device__ __forceinline__ unsigned pkbf(float lo, float hi) {
    v2bf v;
    v[0] = (__bf16)lo;
    v[1] = (__bf16)hi;
    return __builtin_bit_cast(unsigned, v);
}

// One workgroup: (pair p, batch b, 128-row block). 8 waves, one 16-row tile each.
// Streams 16-column target tiles through double-buffered LDS in B-fragment layout;
// flash-style single-exp online logsumexp (base-2); diagonal captured in the one
// tile where colbase == rowbase; partial row-loss sums to a private ws slot.
__global__ __launch_bounds__(256)
void tics_rowblock_kernel(const float* __restrict__ q1, const float* __restrict__ k2,
                          const float* __restrict__ q2, const float* __restrict__ k1,
                          const int* __restrict__ lq1, const int* __restrict__ lk2,
                          const int* __restrict__ lq2, const int* __restrict__ lk1,
                          float* __restrict__ ws) {
    const int tid  = threadIdx.x;
    const int wave = tid >> 5;
    const int lane = tid & 31;
    const int b    = blockIdx.y;
    const int p    = blockIdx.z;
    const int rb   = blockIdx.x;

    const float* pred   = (p == 0) ? q1 : q2;
    const float* target = (p == 0) ? k2 : k1;
    const int lp = (p == 0) ? lq1[b] : lq2[b];
    const int lt = (p == 0) ? lk2[b] : lk1[b];
    const int minl = lp < lt ? lp : lt;

    float* slot = ws + ((p * B_DIM + b) * NBLK + rb);
    const int rowbase_blk = rb * ROWS_PER_BLOCK;
    if (minl <= 0 || rowbase_blk >= minl) {
        if (tid == 0) *slot = 0.0f;
        return;
    }

    __shared__ __align__(64) unsigned short bt[2][16 * 256];  // double-buffered 16x256 bf16 tiles
    __shared__ float wsum;
    if (tid == 0) wsum = 0.0f;

    const int m = lane & 15;       // A row index / C column index within tile
    const int h = lane >> 4;       // half-wave
    const int rowbase = rowbase_blk + wave * 16;

    // ---- Load this wave's A strip (16 rows x 256 d) into registers as bf16 ----
    // A layout (16-bit 16x32): lane m+16h holds row m, K = 32kc+16h+{0..15}
    const float* arow = pred + (size_t)(rowbase + m) * (B_DIM * D_DIM) + (size_t)b * D_DIM;
    v16bf afrag[8];
#pragma unroll
    for (int kc = 0; kc < 8; ++kc) {
        const float* s1 = arow + 32 * kc + 16 * h;
        float4 f0 = *(const float4*)(s1);
        float4 f1 = *(const float4*)(s1 + 4);
        float4 f2 = *(const float4*)(s1 + 8);
        float4 f3 = *(const float4*)(s1 + 12);
        v8u au;
        au[0] = pkbf(f0.x, f0.y); au[1] = pkbf(f0.z, f0.w);
        au[2] = pkbf(f1.x, f1.y); au[3] = pkbf(f1.z, f1.w);
        au[4] = pkbf(f2.x, f2.y); au[5] = pkbf(f2.z, f2.w);
        au[6] = pkbf(f3.x, f3.y); au[7] = pkbf(f3.z, f3.w);
        afrag[kc] = __builtin_bit_cast(v16bf, au);
    }

    // Online LSE state (base-2) for this lane's 8 rows (rows r+8h), plus diagonal.
    float mr[8], sr[8], dr[8];
#pragma unroll
    for (int r = 0; r < 8; ++r) { mr[r] = NEGINF; sr[r] = 0.0f; dr[r] = 0.0f; }

    // Cooperative loader mapping: thread -> (tile row, 16-d segment).
    const int ldrow = tid >> 4;            // 0..15  (target row within tile = output col n)
    const int ldseg = tid & 15;            // 0..15  (d segment of 16)
    // B-fragment LDS layout: region(kc, L) at ushort offset kc*512 + L*16, L = n + 16*(K half)
    const int ldoff = (ldseg >> 1) * 512 + (ldrow + 16 * (ldseg & 1)) * 16;

    const int ntiles = (minl + 15) >> 4;

    // Loads the 16-float segment of tile `nt`, converts to bf16, packs to 2x uint4.
    auto load_tile = [&](int nt, uint4& u0, uint4& u1) {
        const float* trow = target + (size_t)((nt << 4) + ldrow) * (B_DIM * D_DIM)
                          + (size_t)b * D_DIM + ldseg * 16;
        float4 g0 = *(const float4*)(trow);
        float4 g1 = *(const float4*)(trow + 4);
        float4 g2 = *(const float4*)(trow + 8);
        float4 g3 = *(const float4*)(trow + 12);
        __builtin_prefetch(trow + 16 * (B_DIM * D_DIM), 0, 0);  // next tile -> global_prefetch_b8
        u0.x = pkbf(g0.x, g0.y); u0.y = pkbf(g0.z, g0.w);
        u0.z = pkbf(g1.x, g1.y); u0.w = pkbf(g1.z, g1.w);
        u1.x = pkbf(g2.x, g2.y); u1.y = pkbf(g2.z, g2.w);
        u1.z = pkbf(g3.x, g3.y); u1.w = pkbf(g3.z, g3.w);
    };

    // Preload tile 0 into buffer 0.
    {
        uint4 u0, u1;
        load_tile(0, u0, u1);
        *(uint4*)&bt[0][ldoff]     = u0;
        *(uint4*)&bt[0][ldoff + 8] = u1;
    }

    for (int nt = 0; nt < ntiles; ++nt) {
        __syncthreads();   // bt[nt&1] ready; bt[(nt+1)&1] free (its readers finished last iter)

        // Kick off next tile's global loads + LDS stores before computing this one.
        if (nt + 1 < ntiles) {
            uint4 u0, u1;
            load_tile(nt + 1, u0, u1);
            unsigned short* nbuf = bt[(nt + 1) & 1];
            *(uint4*)&nbuf[ldoff]     = u0;
            *(uint4*)&nbuf[ldoff + 8] = u1;
        }

        const unsigned short* cbuf = bt[nt & 1];
        v8f c = {};
#pragma unroll
        for (int kc = 0; kc < 8; ++kc) {
            v16bf bv = *(const v16bf*)&cbuf[kc * 512 + lane * 16];
            c = __builtin_amdgcn_wmma_f32_16x16x32_bf16(
                    false, afrag[kc], false, bv, (short)0, c, false, false);
        }

        const int colbase = nt << 4;
        float v[8];
        if (colbase + 16 <= minl) {                 // wave-uniform: full tile, no masking
#pragma unroll
            for (int r = 0; r < 8; ++r) v[r] = c[r] * SCALE2;
        } else {
            const bool cvalid = (colbase + m) < minl;
#pragma unroll
            for (int r = 0; r < 8; ++r) v[r] = cvalid ? c[r] * SCALE2 : NEGINF;
        }

        if (colbase == rowbase) {                   // wave-uniform: the only diag-carrying tile
#pragma unroll
            for (int r = 0; r < 8; ++r)
                if (r + 8 * h == m) dr[r] = v[r];
        }

        // Single-exp2 online LSE update: one of 2^(m-nm), 2^(v-nm) is always 1.
#pragma unroll
        for (int r = 0; r < 8; ++r) {
            float nm = fmaxf(mr[r], v[r]);
            float e  = exp2_fast(fminf(mr[r], v[r]) - nm);
            bool  gt = v[r] > mr[r];
            sr[r] = fmaf(sr[r], gt ? e : 1.0f, gt ? 1.0f : e);
            mr[r] = nm;
        }
    }

    // Merge (m,s,diag) across the 16 lanes of each half-wave (single exp2 per merge).
#pragma unroll
    for (int r = 0; r < 8; ++r) {
#pragma unroll
        for (int off = 8; off >= 1; off >>= 1) {
            float om = __shfl_xor(mr[r], off, 16);
            float os = __shfl_xor(sr[r], off, 16);
            float od = __shfl_xor(dr[r], off, 16);
            float nm = fmaxf(mr[r], om);
            float e  = exp2_fast(fminf(mr[r], om) - nm);
            bool  sm = mr[r] >= om;     // self holds the max
            sr[r] = fmaf(sm ? os : sr[r], e, sm ? sr[r] : os);
            mr[r] = nm;
            dr[r] += od;
        }
    }

    float local = 0.0f;
    if (m == 0) {  // lanes 0 and 16 of each wave hold the fully-reduced rows
#pragma unroll
        for (int r = 0; r < 8; ++r) {
            const int rowg = rowbase + r + 8 * h;
            if (rowg < minl)            // convert base-2 LSE back to natural log
                local += (mr[r] + log2_fast(sr[r]) - dr[r]) * LN2;
        }
    }
    __syncthreads();                       // wsum == 0.0 visible; all compute done
    if (m == 0) atomicAdd(&wsum, local);
    __syncthreads();
    if (tid == 0) *slot = wsum;
}

// Single small block: per-(pair,batch) normalization + valid-batch averaging.
__global__ void tics_finalize_kernel(const int* __restrict__ lq1, const int* __restrict__ lk2,
                                     const int* __restrict__ lq2, const int* __restrict__ lk1,
                                     const float* __restrict__ ws, float* __restrict__ out) {
    __shared__ float tot[2];
    __shared__ float nvv[2];
    const int t = threadIdx.x;         // 32 threads: (pair, batch)
    const int p = t >> 4, b = t & 15;
    const int lp = (p == 0) ? lq1[b] : lq2[b];
    const int lt = (p == 0) ? lk2[b] : lk1[b];
    const int minl = lp < lt ? lp : lt;
    float sum = 0.0f;
    for (int r = 0; r < NBLK; ++r)
        sum += ws[(p * B_DIM + b) * NBLK + r];
    float per = (minl > 0) ? sum / (float)minl : 0.0f;
    float val = (minl > 0) ? 1.0f : 0.0f;
    for (int off = 8; off >= 1; off >>= 1) {
        per += __shfl_xor(per, off, 16);
        val += __shfl_xor(val, off, 16);
    }
    if (b == 0) { tot[p] = per; nvv[p] = val; }
    __syncthreads();
    if (t == 0) {
        float la = (nvv[0] > 0.0f) ? tot[0] / nvv[0] : 0.0f;
        float lb = (nvv[1] > 0.0f) ? tot[1] / nvv[1] : 0.0f;
        out[0] = 0.5f * (la + lb);
    }
}

extern "C" void kernel_launch(void* const* d_in, const int* in_sizes, int n_in,
                              void* d_out, int out_size, void* d_ws, size_t ws_size,
                              hipStream_t stream) {
    const float* q1 = (const float*)d_in[0];
    const float* k2 = (const float*)d_in[1];
    const float* q2 = (const float*)d_in[2];
    const float* k1 = (const float*)d_in[3];
    const int* lq1 = (const int*)d_in[6];
    const int* lk2 = (const int*)d_in[7];
    const int* lq2 = (const int*)d_in[8];
    const int* lk1 = (const int*)d_in[9];
    float* ws = (float*)d_ws;   // 2 * 16 * NBLK floats; every slot written each call

    dim3 grid(NBLK, B_DIM, 2);
    hipLaunchKernelGGL(tics_rowblock_kernel, grid, dim3(256), 0, stream,
                       q1, k2, q2, k1, lq1, lk2, lq2, lk1, ws);
    hipLaunchKernelGGL(tics_finalize_kernel, dim3(1), dim3(32), 0, stream,
                       lq1, lk2, lq2, lk1, ws, (float*)d_out);
}
